// MoELayer_78855599554933
// MI455X (gfx1250) — matrix operand order repaired and checked by reading, not compiled, not run on hardware
//
#include <hip/hip_runtime.h>
#include <hip/hip_bf16.h>
#include <stdint.h>

#define N_TOK 2048
#define DDIM  1024
#define IDIM  1024
#define NEXP  8
#define NMAT  9   // 8 routed + 1 shared

typedef __attribute__((ext_vector_type(16))) __bf16 v16bf;
typedef __attribute__((ext_vector_type(2)))  __bf16 v2bf;
typedef __attribute__((ext_vector_type(8)))  float  v8f;
typedef int v4i_vs __attribute__((__vector_size__(16)));

union AFrag { v16bf v; uint4 q[2]; };

// native f32 -> bf16 (RNE); backend may select v_cvt_pk_bf16_f32
__device__ inline unsigned short cvt_bf16(float a) {
    union { __bf16 h; unsigned short s; } c;
    c.h = (__bf16)a;
    return c.s;
}

__device__ inline uint32_t pack_bf16x2(float a, float b) {
    union { v2bf v; uint32_t u; } c;
    c.v.x = (__bf16)a;
    c.v.y = (__bf16)b;
    return c.u;
}

__device__ inline float fast_rcp(float x) {
#if __has_builtin(__builtin_amdgcn_rcpf)
    return __builtin_amdgcn_rcpf(x);
#else
    return 1.0f / x;
#endif
}

// ---- CDNA5 async global->LDS copy (ASYNCcnt), with safe fallback ----
#if __has_builtin(__builtin_amdgcn_global_load_async_to_lds_b128)
#define HAVE_ASYNC_LDS 1
#endif

__device__ inline void async_copy16(const unsigned short* g, unsigned short* l) {
#ifdef HAVE_ASYNC_LDS
    __builtin_amdgcn_global_load_async_to_lds_b128(
        (__attribute__((address_space(1))) v4i_vs*)(void*)g,
        (__attribute__((address_space(3))) v4i_vs*)(void*)l, 0, 0);
#else
    *reinterpret_cast<uint4*>(l) = *reinterpret_cast<const uint4*>(g);
#endif
}

__device__ inline void wait_async_lds() {
#ifdef HAVE_ASYNC_LDS
#if __has_builtin(__builtin_amdgcn_s_wait_asynccnt)
    __builtin_amdgcn_s_wait_asynccnt(0);
#else
    asm volatile("s_wait_asynccnt 0x0" ::: "memory");
#endif
#endif
}

// ---------------- aux init ----------------
__global__ void init_aux_kernel(float* aux) {
    if (threadIdx.x < 16) aux[threadIdx.x] = 0.0f;
}

// ---------------- router ----------------
__global__ __launch_bounds__(256) void router_kernel(
    const float* __restrict__ x, const float* __restrict__ wgate,
    float* __restrict__ cw, float* __restrict__ aux) {
    const int lane = threadIdx.x & 31;
    const int wave = threadIdx.x >> 5;
    const int n = blockIdx.x * 8 + wave;

    float p[NEXP];
    #pragma unroll
    for (int e = 0; e < NEXP; ++e) p[e] = 0.0f;

    for (int d = lane; d < DDIM; d += 32) {
        float xv = x[(size_t)n * DDIM + d];
        const float* wr = wgate + (size_t)d * NEXP;
        #pragma unroll
        for (int e = 0; e < NEXP; ++e) p[e] += xv * wr[e];
    }
    #pragma unroll
    for (int off = 16; off > 0; off >>= 1)
        #pragma unroll
        for (int e = 0; e < NEXP; ++e) p[e] += __shfl_down(p[e], off, 32);

    if (lane == 0) {
        int i0 = 0; float l0 = p[0];
        #pragma unroll
        for (int e = 1; e < NEXP; ++e) if (p[e] > l0) { l0 = p[e]; i0 = e; }
        int i1 = -1; float l1 = -3.0e38f;
        #pragma unroll
        for (int e = 0; e < NEXP; ++e) if (e != i0 && p[e] > l1) { l1 = p[e]; i1 = e; }
        float w0 = 1.0f / (1.0f + __expf(l1 - l0));
        float w1 = 1.0f - w0;
        #pragma unroll
        for (int e = 0; e < NEXP; ++e)
            cw[(size_t)n * NEXP + e] = (e == i0) ? w0 : ((e == i1) ? w1 : 0.0f);
        float s = 0.0f, ex[NEXP];
        #pragma unroll
        for (int e = 0; e < NEXP; ++e) { ex[e] = __expf(p[e] - l0); s += ex[e]; }
        float inv = 1.0f / s;
        #pragma unroll
        for (int e = 0; e < NEXP; ++e) atomicAdd(&aux[8 + e], ex[e] * inv);
        atomicAdd(&aux[i0], 1.0f);
        atomicAdd(&aux[i1], 1.0f);
    }
}

// ---------------- mlp1: act'[e] = cw[:,e] * silu(x@Wg) * (x@Wu) ----------------
__global__ __launch_bounds__(256) void mlp1_kernel(
    const float* __restrict__ x,
    const float* __restrict__ wg, const float* __restrict__ wu,
    const float* __restrict__ wsg, const float* __restrict__ wsu,
    const float* __restrict__ cw, unsigned short* __restrict__ actw) {
    const int e = blockIdx.z;
    const int rowBase = blockIdx.y * 128;
    const int colBase = blockIdx.x * 128;
    const float* Wg = (e < NEXP) ? (wg + (size_t)e * DDIM * IDIM) : wsg;
    const float* Wu = (e < NEXP) ? (wu + (size_t)e * DDIM * IDIM) : wsu;

    __shared__ alignas(16) unsigned short Al[2][128 * 40];
    __shared__ alignas(16) unsigned short Bg[2][128 * 40];
    __shared__ alignas(16) unsigned short Bu[2][128 * 40];
    __shared__ float cwl[128];

    const int t = threadIdx.x;
    if (t < 128) cwl[t] = (e < NEXP) ? cw[(size_t)(rowBase + t) * NEXP + e] : 1.0f;

    const int lane  = t & 31;
    const int wave  = t >> 5;
    const int waveM = (wave >> 2) * 64;
    const int waveN = (wave & 3) * 32;

    v8f accg[4][2], accu[4][2];
    const v8f vzero = {0.f,0.f,0.f,0.f,0.f,0.f,0.f,0.f};
    #pragma unroll
    for (int i = 0; i < 4; ++i)
        #pragma unroll
        for (int j = 0; j < 2; ++j) { accg[i][j] = vzero; accu[i][j] = vzero; }

    const int NIT = DDIM / 32;   // 32
    float4 ra[4], rg[4], ru[4];

    auto loadAll = [&](int it) {
        const int kb = it * 32;
        #pragma unroll
        for (int i = 0; i < 4; ++i) {
            int idx = t + i * 256;           // A: 0..1023
            int r = idx >> 3, cq = idx & 7;
            ra[i] = *reinterpret_cast<const float4*>(
                x + (size_t)(rowBase + r) * DDIM + kb + cq * 4);
        }
        #pragma unroll
        for (int i = 0; i < 4; ++i) {
            int idx = t + i * 256;
            int cq = idx & 31, kr = idx >> 5;
            rg[i] = *reinterpret_cast<const float4*>(
                Wg + (size_t)(kb + kr) * IDIM + colBase + cq * 4);
            ru[i] = *reinterpret_cast<const float4*>(
                Wu + (size_t)(kb + kr) * IDIM + colBase + cq * 4);
        }
        if (it + 2 < NIT) {
            int kb2 = (it + 2) * 32;
            __builtin_prefetch(Wg + (size_t)(kb2 + (t >> 3)) * IDIM + colBase + (t & 7) * 16, 0, 3);
            __builtin_prefetch(Wu + (size_t)(kb2 + (t >> 3)) * IDIM + colBase + (t & 7) * 16, 0, 3);
        }
    };
    auto storeAll = [&](int b) {
        #pragma unroll
        for (int i = 0; i < 4; ++i) {
            int idx = t + i * 256;
            int r = idx >> 3, cq = idx & 7;
            uint32_t* dst = reinterpret_cast<uint32_t*>(&Al[b][r * 40 + cq * 4]);
            dst[0] = pack_bf16x2(ra[i].x, ra[i].y);
            dst[1] = pack_bf16x2(ra[i].z, ra[i].w);
        }
        #pragma unroll
        for (int i = 0; i < 4; ++i) {
            int idx = t + i * 256;
            int cq = idx & 31, kr = idx >> 5;
            Bg[b][(cq * 4 + 0) * 40 + kr] = cvt_bf16(rg[i].x);
            Bg[b][(cq * 4 + 1) * 40 + kr] = cvt_bf16(rg[i].y);
            Bg[b][(cq * 4 + 2) * 40 + kr] = cvt_bf16(rg[i].z);
            Bg[b][(cq * 4 + 3) * 40 + kr] = cvt_bf16(rg[i].w);
            Bu[b][(cq * 4 + 0) * 40 + kr] = cvt_bf16(ru[i].x);
            Bu[b][(cq * 4 + 1) * 40 + kr] = cvt_bf16(ru[i].y);
            Bu[b][(cq * 4 + 2) * 40 + kr] = cvt_bf16(ru[i].z);
            Bu[b][(cq * 4 + 3) * 40 + kr] = cvt_bf16(ru[i].w);
        }
    };

    // pipeline prologue
    loadAll(0);
    storeAll(0);
    if (NIT > 1) loadAll(1);
    __syncthreads();

    for (int it = 0; it < NIT; ++it) {
        const int cur = it & 1, nxt = cur ^ 1;
        if (it + 1 < NIT) {
            storeAll(nxt);                  // regs currently hold tile it+1
            if (it + 2 < NIT) loadAll(it + 2);
        }

        AFrag af[4], fg[2], fu[2];
        const int akk = (lane < 16) ? 0 : 8;
        #pragma unroll
        for (int mt = 0; mt < 4; ++mt) {
            int row = waveM + mt * 16 + (lane & 15);
            af[mt].q[0] = *reinterpret_cast<const uint4*>(&Al[cur][row * 40 + akk]);
            af[mt].q[1] = *reinterpret_cast<const uint4*>(&Al[cur][row * 40 + akk + 16]);
        }
        const int bkk = (lane < 16) ? 0 : 8;
        #pragma unroll
        for (int nt = 0; nt < 2; ++nt) {
            int col = waveN + nt * 16 + (lane & 15);
            fg[nt].q[0] = *reinterpret_cast<const uint4*>(&Bg[cur][col * 40 + bkk]);
            fg[nt].q[1] = *reinterpret_cast<const uint4*>(&Bg[cur][col * 40 + bkk + 16]);
            fu[nt].q[0] = *reinterpret_cast<const uint4*>(&Bu[cur][col * 40 + bkk]);
            fu[nt].q[1] = *reinterpret_cast<const uint4*>(&Bu[cur][col * 40 + bkk + 16]);
        }
        #pragma unroll
        for (int mt = 0; mt < 4; ++mt)
            #pragma unroll
            for (int nt = 0; nt < 2; ++nt) {
                accg[mt][nt] = __builtin_amdgcn_wmma_f32_16x16x32_bf16(
                    false, af[mt].v, false, fg[nt].v, (short)0, accg[mt][nt], false, false);
                accu[mt][nt] = __builtin_amdgcn_wmma_f32_16x16x32_bf16(
                    false, af[mt].v, false, fu[nt].v, (short)0, accu[mt][nt], false, false);
            }
        __syncthreads();
    }

    // epilogue: act' = cw * silu(g) * u -> bf16 ws
    #pragma unroll
    for (int mt = 0; mt < 4; ++mt)
        #pragma unroll
        for (int nt = 0; nt < 2; ++nt) {
            int lrow0 = waveM + mt * 16 + ((lane >> 4) << 3);
            int gcol  = colBase + waveN + nt * 16 + (lane & 15);
            #pragma unroll
            for (int r = 0; r < 8; ++r) {
                int lrow = lrow0 + r;
                float g = accg[mt][nt][r];
                float u = accu[mt][nt][r];
                float act = g * fast_rcp(1.0f + __expf(-g)) * u * cwl[lrow];
                actw[(size_t)e * N_TOK * IDIM + (size_t)(rowBase + lrow) * IDIM + gcol]
                    = cvt_bf16(act);
            }
        }
}

// ---------------- mlp2: out = sum_e act'[e] @ Wd[e] (K = 9*1024) ----------------
__global__ __launch_bounds__(256) void mlp2_kernel(
    const unsigned short* __restrict__ actw,
    const float* __restrict__ wd, const float* __restrict__ wsd,
    float* __restrict__ out) {
    const int rowBase = blockIdx.y * 128;
    const int colBase = blockIdx.x * 128;

    __shared__ alignas(16) unsigned short Al[2][128 * 40];
    __shared__ alignas(16) unsigned short Bt[2][128 * 40];

    const int t = threadIdx.x;
    const int lane  = t & 31;
    const int wave  = t >> 5;
    const int waveM = (wave >> 2) * 64;
    const int waveN = (wave & 3) * 32;

    v8f acc[4][2];
    const v8f vzero = {0.f,0.f,0.f,0.f,0.f,0.f,0.f,0.f};
    #pragma unroll
    for (int i = 0; i < 4; ++i)
        #pragma unroll
        for (int j = 0; j < 2; ++j) acc[i][j] = vzero;

    const int NIT = NMAT * (IDIM / 32);   // 288
    float4 rb[4];

    auto issueA = [&](int it, int b) {     // async bf16 tile copy: global -> LDS
        const int e  = it >> 5;
        const int kb = (it & 31) << 5;
        const unsigned short* ap = actw + (size_t)e * N_TOK * IDIM + kb;
        #pragma unroll
        for (int i = 0; i < 2; ++i) {
            int idx = t + i * 256;          // 512 x uint4
            int r = idx >> 2, q = idx & 3;
            async_copy16(ap + (size_t)(rowBase + r) * IDIM + q * 8,
                         &Al[b][r * 40 + q * 8]);
        }
    };
    auto loadB = [&](int it) {
        const int e  = it >> 5;
        const int kb = (it & 31) << 5;
        const float* wp = (e < NEXP) ? (wd + (size_t)e * IDIM * DDIM) : wsd;
        #pragma unroll
        for (int i = 0; i < 4; ++i) {
            int idx = t + i * 256;
            int cq = idx & 31, kr = idx >> 5;
            rb[i] = *reinterpret_cast<const float4*>(
                wp + (size_t)(kb + kr) * DDIM + colBase + cq * 4);
        }
        __builtin_prefetch(wp + (size_t)(kb + 32 + (t >> 3)) * DDIM + colBase + (t & 7) * 16, 0, 3);
    };
    auto storeB = [&](int b) {
        #pragma unroll
        for (int i = 0; i < 4; ++i) {
            int idx = t + i * 256;
            int cq = idx & 31, kr = idx >> 5;
            Bt[b][(cq * 4 + 0) * 40 + kr] = cvt_bf16(rb[i].x);
            Bt[b][(cq * 4 + 1) * 40 + kr] = cvt_bf16(rb[i].y);
            Bt[b][(cq * 4 + 2) * 40 + kr] = cvt_bf16(rb[i].z);
            Bt[b][(cq * 4 + 3) * 40 + kr] = cvt_bf16(rb[i].w);
        }
    };

    // pipeline prologue
    issueA(0, 0);
    loadB(0);
    storeB(0);
    if (NIT > 1) loadB(1);
    wait_async_lds();
    __syncthreads();

    for (int it = 0; it < NIT; ++it) {
        const int cur = it & 1, nxt = cur ^ 1;
        if (it + 1 < NIT) {
            issueA(it + 1, nxt);            // ASYNCcnt-tracked DMA into next buffer
            storeB(nxt);                    // regs hold tile it+1
            if (it + 2 < NIT) loadB(it + 2);
        }

        AFrag af[4], bfg[2];
        const int akk = (lane < 16) ? 0 : 8;
        #pragma unroll
        for (int mt = 0; mt < 4; ++mt) {
            int row = waveM + mt * 16 + (lane & 15);
            af[mt].q[0] = *reinterpret_cast<const uint4*>(&Al[cur][row * 40 + akk]);
            af[mt].q[1] = *reinterpret_cast<const uint4*>(&Al[cur][row * 40 + akk + 16]);
        }
        const int bkk = (lane < 16) ? 0 : 8;
        #pragma unroll
        for (int nt = 0; nt < 2; ++nt) {
            int col = waveN + nt * 16 + (lane & 15);
            bfg[nt].q[0] = *reinterpret_cast<const uint4*>(&Bt[cur][col * 40 + bkk]);
            bfg[nt].q[1] = *reinterpret_cast<const uint4*>(&Bt[cur][col * 40 + bkk + 16]);
        }
        #pragma unroll
        for (int mt = 0; mt < 4; ++mt)
            #pragma unroll
            for (int nt = 0; nt < 2; ++nt)
                acc[mt][nt] = __builtin_amdgcn_wmma_f32_16x16x32_bf16(
                    false, af[mt].v, false, bfg[nt].v, (short)0, acc[mt][nt], false, false);

        wait_async_lds();                   // own async (it+1) landed
        __syncthreads();                    // everyone's landed
    }

    #pragma unroll
    for (int mt = 0; mt < 4; ++mt)
        #pragma unroll
        for (int nt = 0; nt < 2; ++nt) {
            int lrow0 = waveM + mt * 16 + ((lane >> 4) << 3);
            int gcol  = colBase + waveN + nt * 16 + (lane & 15);
            #pragma unroll
            for (int r = 0; r < 8; ++r)
                out[(size_t)(rowBase + lrow0 + r) * DDIM + gcol] = acc[mt][nt][r];
        }
}

// ---------------- finalize aux loss ----------------
__global__ void finalize_kernel(const float* __restrict__ aux, float* __restrict__ out) {
    if (threadIdx.x == 0 && blockIdx.x == 0) {
        float s = 0.0f;
        #pragma unroll
        for (int e = 0; e < NEXP; ++e)
            s += (aux[e] / (float)N_TOK) * (aux[8 + e] / (float)N_TOK);
        out[(size_t)N_TOK * DDIM] = (float)NEXP * s;
    }
}

extern "C" void kernel_launch(void* const* d_in, const int* in_sizes, int n_in,
                              void* d_out, int out_size, void* d_ws, size_t ws_size,
                              hipStream_t stream) {
    const float* x     = (const float*)d_in[0];
    const float* wgate = (const float*)d_in[1];
    const float* wg    = (const float*)d_in[2];
    const float* wu    = (const float*)d_in[3];
    const float* wd    = (const float*)d_in[4];
    const float* wsg   = (const float*)d_in[5];
    const float* wsu   = (const float*)d_in[6];
    const float* wsd   = (const float*)d_in[7];
    float* out = (float*)d_out;

    unsigned short* actw = (unsigned short*)d_ws;                      // 9*2048*1024 bf16
    size_t act_bytes = (size_t)NMAT * N_TOK * IDIM * sizeof(unsigned short);
    float* cw  = (float*)((char*)d_ws + act_bytes);                    // 2048*8 f32
    float* aux = (float*)((char*)d_ws + act_bytes + (size_t)N_TOK * NEXP * sizeof(float)); // 16 f32

    init_aux_kernel<<<1, 32, 0, stream>>>(aux);
    router_kernel<<<N_TOK / 8, 256, 0, stream>>>(x, wgate, cw, aux);
    mlp1_kernel<<<dim3(IDIM / 128, N_TOK / 128, NMAT), 256, 0, stream>>>(
        x, wg, wu, wsg, wsu, cw, actw);
    mlp2_kernel<<<dim3(DDIM / 128, N_TOK / 128, 1), 256, 0, stream>>>(
        actw, wd, wsd, out);
    finalize_kernel<<<1, 32, 0, stream>>>(aux, out);
}
